// CNNEmbedding_38242388803834
// MI455X (gfx1250) — matrix operand order repaired and checked by reading, not compiled, not run on hardware
//
#include <hip/hip_runtime.h>
#include <hip/hip_bf16.h>

typedef __attribute__((ext_vector_type(16))) __bf16 v16bf;
typedef __attribute__((ext_vector_type(8)))  float  v8f;
typedef __attribute__((ext_vector_type(4)))  unsigned int u32x4;
typedef __attribute__((ext_vector_type(8)))  unsigned int u32x8;

union Frag {
    v16bf v;
    uint4 q[2];
};

__device__ __forceinline__ unsigned short f2bf(float f) {
    unsigned u = __builtin_bit_cast(unsigned, f);
    unsigned r = (u + 0x7FFFu + ((u >> 16) & 1u)) >> 16;   // round-to-nearest-even
    return (unsigned short)r;
}

// Cross-half (lane ^ 16) swap as VALU (v_permlanex16_b32, identity selects).
__device__ __forceinline__ float xor16(float x) {
    int v = __builtin_bit_cast(int, x);
    int r = __builtin_amdgcn_permlanex16(v, v, 0x76543210, 0xfedcba98, false, false);
    return __builtin_bit_cast(float, r);
}

// ---------------------------------------------------------------------------
// Prepack: Wflat[r=go + k*16 + o][e] = w_g[o][e][k] in bf16, row-major (128/row)
// Groups (K=2..6) at row offsets {0,32,80,144,224}, 320 rows total.
// ---------------------------------------------------------------------------
__global__ void cnn_prepack(const float* __restrict__ w2, const float* __restrict__ w3,
                            const float* __restrict__ w4, const float* __restrict__ w5,
                            const float* __restrict__ w6, unsigned short* __restrict__ wflat) {
    int tid = blockIdx.x * blockDim.x + threadIdx.x;
    if (tid >= 320 * 128) return;
    int r = tid >> 7;
    int e = tid & 127;
    int g, go, K;
    if      (r <  32) { g = 0; go =   0; K = 2; }
    else if (r <  80) { g = 1; go =  32; K = 3; }
    else if (r < 144) { g = 2; go =  80; K = 4; }
    else if (r < 224) { g = 3; go = 144; K = 5; }
    else              { g = 4; go = 224; K = 6; }
    int rr = r - go;
    int k = rr >> 4;
    int o = rr & 15;
    const float* W[5] = {w2, w3, w4, w5, w6};
    float v = W[g][(o * 128 + e) * K + k];
    wflat[r * 128 + e] = f2bf(v);
}

// ---------------------------------------------------------------------------
// Per-group conv + maxpool.
// D[t, r] tiles: lane holds filter column o = lane&15; t in VGPRs/lane-halves.
// val[] is built in LOW-HALF semantics only (no per-lane selects); high lanes
// compute a permuted/garbage conv+max that is never stored (store is lane<16).
// ---------------------------------------------------------------------------
template <int KK, int P, int GO, int G>
__device__ __forceinline__ void conv_group(const Frag* A, const unsigned short* Wl,
                                           int lane, bool lohalf,
                                           const float* __restrict__ bias,
                                           float* __restrict__ outw) {
    constexpr int LOUT = 32 + 2 * P - KK + 1;
    float conv[LOUT];
#pragma unroll
    for (int t = 0; t < LOUT; ++t) conv[t] = 0.f;

    int rcol   = GO + (lane & 15);
    int ebhalf = lohalf ? 0 : 8;

#pragma unroll
    for (int k = 0; k < KK; ++k) {
        v8f acc0 = {0.f, 0.f, 0.f, 0.f, 0.f, 0.f, 0.f, 0.f};
        v8f acc1 = {0.f, 0.f, 0.f, 0.f, 0.f, 0.f, 0.f, 0.f};
        const unsigned short* wrow = Wl + (rcol + 16 * k) * 128 + ebhalf;
#pragma unroll
        for (int ks = 0; ks < 4; ++ks) {
            Frag bf;
            const uint4* bp = (const uint4*)(wrow + ks * 32);   // e = eb..eb+7, eb+16..eb+23
            bf.q[0] = bp[0];
            bf.q[1] = bp[2];
            acc0 = __builtin_amdgcn_wmma_f32_16x16x32_bf16(false, A[ks].v,     false, bf.v,
                                                           (short)0, acc0, false, false);
            acc1 = __builtin_amdgcn_wmma_f32_16x16x32_bf16(false, A[4 + ks].v, false, bf.v,
                                                           (short)0, acc1, false, false);
        }
        // Low-half canonical order: t=i from own acc0, t=8+i from partner half,
        // t=16+i from own acc1, t=24+i from partner half. No cndmask selects.
        float val[32];
#pragma unroll
        for (int i = 0; i < 8; ++i) {
            val[i]      = acc0[i];
            val[16 + i] = acc1[i];
        }
#pragma unroll
        for (int i = 0; i < 8; ++i) {
            val[8 + i]  = xor16(acc0[i]);
            val[24 + i] = xor16(acc1[i]);
        }
#pragma unroll
        for (int t = 0; t < LOUT; ++t) {
            int ti = t - P + k;
            if (ti >= 0 && ti < 32) conv[t] += val[ti];   // zero-pad taps fold out
        }
    }
    float m = conv[0];
#pragma unroll
    for (int t = 1; t < LOUT; ++t) m = fmaxf(m, conv[t]);
    if (lane < 16) outw[G * 16 + lane] = m + bias[lane];
}

// ---------------------------------------------------------------------------
// Main kernel: 8 waves/block, one word per wave.
// ---------------------------------------------------------------------------
#define NW 8

__global__ __launch_bounds__(256) void cnn_main(
    const int* __restrict__ word, const float* __restrict__ emb,
    const unsigned short* __restrict__ wflat,
    const float* __restrict__ b2, const float* __restrict__ b3,
    const float* __restrict__ b4, const float* __restrict__ b5,
    const float* __restrict__ b6, float* __restrict__ out) {

    __shared__ __align__(16) unsigned short Wlds[320 * 128];      // 80 KB bf16 Wflat
    __shared__ __align__(16) unsigned short Xlds[NW][32 * 128];   // 8 KB/wave gathered panel

    const int lane = threadIdx.x & 31;
    const int wv   = threadIdx.x >> 5;
    const bool lohalf = lane < 16;

    // --- Stage Wflat (80 KB) into LDS with the Tensor Data Mover: one 2D
    // descriptor, tile = tensor = 128 x 320 elements of 2 bytes, contiguous.
    if (wv == 0) {
        unsigned long long ga = (unsigned long long)(uintptr_t)wflat;
        unsigned ldsa = (unsigned)(uintptr_t)(&Wlds[0]);
        u32x4 g0;
        g0[0] = 1u;                                    // count=1, user descriptor
        g0[1] = ldsa;                                  // lds_addr
        g0[2] = (unsigned)ga;                          // global_addr[31:0]
        g0[3] = (unsigned)(ga >> 32) | (2u << 30);     // global_addr[56:32] | type=2
        u32x8 g1;
        g1[0] = 0x00010000u;                           // data_size=2B, no mask/flags
        g1[1] = 128u << 16;                            // tensor_dim0 = 128
        g1[2] = 320u << 16;                            // tensor_dim1 = 320
        g1[3] = 128u << 16;                            // tile_dim0 = 128
        g1[4] = 320u;                                  // tile_dim1 = 320, tile_dim2 = 0
        g1[5] = 128u;                                  // tensor_dim0_stride = 128
        g1[6] = 0u;
        g1[7] = 0u;
        asm volatile("tensor_load_to_lds %0, %1" :: "s"(g0), "s"(g1) : "memory");
    }

    // --- Gather + fp32->bf16 convert this wave's word panel (overlaps the TDM):
    // X[t][e], column t contiguous (256 B), bf16.
    const int b = blockIdx.x * NW + wv;
    const int widx = word[b * 32 + lane];
    unsigned short* Xw = &Xlds[wv][0];
#pragma unroll 4
    for (int t = 0; t < 32; ++t) {
        int idx = __shfl(widx, t, 32);
        float4 v = make_float4(0.f, 0.f, 0.f, 0.f);
        if (idx >= 0) v = *(const float4*)(emb + (size_t)idx * 128 + lane * 4);
        uint2 pk;
        pk.x = (unsigned)f2bf(v.x) | ((unsigned)f2bf(v.y) << 16);
        pk.y = (unsigned)f2bf(v.z) | ((unsigned)f2bf(v.w) << 16);
        *(uint2*)(Xw + t * 128 + lane * 4) = pk;
    }
    if (wv == 0) __builtin_amdgcn_s_wait_tensorcnt(0);
    __syncthreads();

    // A fragments = X^T tiles (M=t, K=e): lane l row t = mt*16+(l&15);
    // lanes>=16 hold K-offset +8 (ISA 16-bit A layout). 2 t-tiles x 4 k-steps.
    Frag A[8];
#pragma unroll
    for (int mt = 0; mt < 2; ++mt) {
#pragma unroll
        for (int ks = 0; ks < 4; ++ks) {
            int t  = mt * 16 + (lane & 15);
            int eb = ks * 32 + (lohalf ? 0 : 8);
            const uint4* ap = (const uint4*)(Xw + t * 128 + eb);
            A[mt * 4 + ks].q[0] = ap[0];   // e = eb .. eb+7      -> VGPRs 0-3
            A[mt * 4 + ks].q[1] = ap[2];   // e = eb+16 .. eb+23  -> VGPRs 4-7
        }
    }

    float* outw = out + (size_t)b * 80;
    conv_group<2, 0,   0, 0>(A, Wlds, lane, lohalf, b2, outw);
    conv_group<3, 0,  32, 1>(A, Wlds, lane, lohalf, b3, outw);
    conv_group<4, 1,  80, 2>(A, Wlds, lane, lohalf, b4, outw);
    conv_group<5, 2, 144, 3>(A, Wlds, lane, lohalf, b5, outw);
    conv_group<6, 3, 224, 4>(A, Wlds, lane, lohalf, b6, outw);
}

extern "C" void kernel_launch(void* const* d_in, const int* in_sizes, int n_in,
                              void* d_out, int out_size, void* d_ws, size_t ws_size,
                              hipStream_t stream) {
    const int*   word = (const int*)d_in[0];
    const float* emb  = (const float*)d_in[1];
    const float* w2 = (const float*)d_in[2];  const float* b2 = (const float*)d_in[3];
    const float* w3 = (const float*)d_in[4];  const float* b3 = (const float*)d_in[5];
    const float* w4 = (const float*)d_in[6];  const float* b4 = (const float*)d_in[7];
    const float* w5 = (const float*)d_in[8];  const float* b5 = (const float*)d_in[9];
    const float* w6 = (const float*)d_in[10]; const float* b6 = (const float*)d_in[11];

    unsigned short* wflat = (unsigned short*)d_ws;   // 320*128 bf16 = 80 KB

    cnn_prepack<<<(320 * 128 + 255) / 256, 256, 0, stream>>>(w2, w3, w4, w5, w6, wflat);

    const int B = 16384;
    cnn_main<<<B / NW, 256, 0, stream>>>(word, emb, wflat, b2, b3, b4, b5, b6, (float*)d_out);
}
